// KascadeAnchorAttention_27204322853612
// MI455X (gfx1250) — matrix operand order repaired and checked by reading, not compiled, not run on hardware
//
#include <hip/hip_runtime.h>

// ---------------------------------------------------------------------------
// Types for CDNA5 WMMA (wave32, 16x16x32 bf16 -> f32)
// ---------------------------------------------------------------------------
typedef __bf16 v16bf __attribute__((ext_vector_type(16)));
typedef float  v8f   __attribute__((ext_vector_type(8)));
typedef int    v4i   __attribute__((vector_size(16)));   // matches builtin param
typedef unsigned short u16;

union ABFrag { v16bf v; uint4 u[2]; u16 s[16]; };

#define NUM_HEADS 16
#define HEAD_DIM  64
#define SEQ       2048
#define BATCH     2
#define MODEL     1024   // NUM_HEADS * HEAD_DIM
#define NTILES    128    // SEQ / 16
#define TOPK      8
#define NEGV      (-1.0e10f)

#define LDT 40           // LDS row stride (u16) for 32-wide tiles: 80B, padded

__device__ __forceinline__ u16 f32_to_bf16(float f) {
    unsigned u = __float_as_uint(f);
    u += 0x7fffu + ((u >> 16) & 1u);      // round-to-nearest-even
    return (u16)(u >> 16);
}
__device__ __forceinline__ float bf16_to_f32(u16 h) {
    return __uint_as_float(((unsigned)h) << 16);
}

// ---------------------------------------------------------------------------
// Async global->LDS copy (16B per lane-op), guarded so a missing builtin
// degrades to plain ld/st staging instead of breaking the build.
// ---------------------------------------------------------------------------
#if __has_builtin(__builtin_amdgcn_global_load_async_to_lds_b128)
#define HAVE_ASYNC_LDS 1
__device__ __forceinline__ void copy16_g2l(const u16* g, u16* l) {
    __builtin_amdgcn_global_load_async_to_lds_b128(
        (__attribute__((address_space(1))) v4i*)(unsigned long long)g,
        (__attribute__((address_space(3))) v4i*)(unsigned)(unsigned long long)l,
        0, 0);
}
__device__ __forceinline__ void wait_async_copies() {
#if __has_builtin(__builtin_amdgcn_s_wait_asynccnt)
    __builtin_amdgcn_s_wait_asynccnt(0);
#else
    asm volatile("s_wait_asynccnt 0" ::: "memory");
#endif
}
#else
#define HAVE_ASYNC_LDS 0
__device__ __forceinline__ void copy16_g2l(const u16* g, u16* l) {
    *(uint4*)l = *(const uint4*)g;
}
__device__ __forceinline__ void wait_async_copies() {}
#endif

// ---------------------------------------------------------------------------
// 1) x fp32 -> bf16
// ---------------------------------------------------------------------------
__global__ __launch_bounds__(256) void cvt_x_kernel(const float* __restrict__ x,
                                                    u16* __restrict__ xb, int n) {
    int i = blockIdx.x * blockDim.x + threadIdx.x;
    if (i < n) xb[i] = f32_to_bf16(x[i]);
}

// ---------------------------------------------------------------------------
// 2) weight fp32 -> bf16, transposed (out[n][k] = in[k][n]).
//    Grid (32,32,4), block (32,8).
// ---------------------------------------------------------------------------
__global__ __launch_bounds__(256) void cvt_w_kernel(
        const float* __restrict__ wq, const float* __restrict__ wk,
        const float* __restrict__ wv, const float* __restrict__ wo,
        u16* __restrict__ wqT, u16* __restrict__ wkT,
        u16* __restrict__ wvT, u16* __restrict__ woT) {
    const float* in; u16* out;
    switch (blockIdx.z) {
        case 0:  in = wq; out = wqT; break;
        case 1:  in = wk; out = wkT; break;
        case 2:  in = wv; out = wvT; break;
        default: in = wo; out = woT; break;
    }
    __shared__ float tile[32][33];
    int bx = blockIdx.x * 32, by = blockIdx.y * 32;
    int tx = threadIdx.x, ty = threadIdx.y;
#pragma unroll
    for (int i = 0; i < 32; i += 8)
        tile[ty + i][tx] = in[(size_t)(by + ty + i) * MODEL + bx + tx];
    __syncthreads();
#pragma unroll
    for (int i = 0; i < 32; i += 8)
        out[(size_t)(bx + ty + i) * MODEL + by + tx] = f32_to_bf16(tile[tx][ty + i]);
}

// ---------------------------------------------------------------------------
// Shared 128x128 GEMM mainloop: C(128x128) += A(128xK) * B^T-layout(128xK),
// K = MODEL, BK = 32. 8 waves in 4x2; each wave computes 32x64 via 8 WMMAs
// per k-step from LDS-staged, double-buffered tiles (async copies).
// ---------------------------------------------------------------------------
__device__ __forceinline__ void stage_tiles(const u16* __restrict__ Ab,
                                            const u16* __restrict__ Bb,
                                            u16* __restrict__ As_,
                                            u16* __restrict__ Bs_,
                                            int tid, int kk) {
    // 128 rows x 32 u16 per tile = 512 16B-chunks; 256 threads x 2 chunks each.
#pragma unroll
    for (int i = 0; i < 2; ++i) {
        int c = tid + i * 256;
        int row = c >> 2;
        int seg = (c & 3) * 8;
        copy16_g2l(Ab + (size_t)row * MODEL + kk + seg, As_ + row * LDT + seg);
        copy16_g2l(Bb + (size_t)row * MODEL + kk + seg, Bs_ + row * LDT + seg);
    }
}

__device__ __forceinline__ void gemm_tile_128(const u16* __restrict__ Ab,
                                              const u16* __restrict__ Bb,
                                              int tid, v8f (&acc)[2][4]) {
    __shared__ u16 As[2][128 * LDT];
    __shared__ u16 Bs[2][128 * LDT];

    const int lane = tid & 31;
    const bool hi  = lane >= 16;
    const int l15  = lane & 15;
    const int wid  = tid >> 5;
    const int wr   = wid >> 1;   // 0..3 (row strip of 32)
    const int wc   = wid & 1;    // 0..1 (col strip of 64)
    const int NSTEP = MODEL / 32;

    stage_tiles(Ab, Bb, As[0], Bs[0], tid, 0);

    for (int s = 0; s < NSTEP; ++s) {
        wait_async_copies();     // my copies for buffer s&1 are done
        __syncthreads();         // everyone's copies done; everyone left step s-1
        if (s + 1 < NSTEP)
            stage_tiles(Ab, Bb, As[(s + 1) & 1], Bs[(s + 1) & 1], tid, (s + 1) * 32);

        const u16* A_ = As[s & 1];
        const u16* B_ = Bs[s & 1];
        ABFrag aF[2];
#pragma unroll
        for (int t = 0; t < 2; ++t) {
            int off = (wr * 32 + t * 16 + l15) * LDT + (hi ? 8 : 0);
            aF[t].u[0] = *(const uint4*)&A_[off];
            aF[t].u[1] = *(const uint4*)&A_[off + 16];
        }
        ABFrag bF[4];
#pragma unroll
        for (int j = 0; j < 4; ++j) {
            int off = (wc * 64 + j * 16 + l15) * LDT + (hi ? 8 : 0);
            bF[j].u[0] = *(const uint4*)&B_[off];
            bF[j].u[1] = *(const uint4*)&B_[off + 16];
        }
#pragma unroll
        for (int t = 0; t < 2; ++t)
#pragma unroll
            for (int j = 0; j < 4; ++j)
                acc[t][j] = __builtin_amdgcn_wmma_f32_16x16x32_bf16(
                    false, aF[t].v, false, bF[j].v, (short)0, acc[t][j], false, false);
    }
}

// ---------------------------------------------------------------------------
// 3) Projections: z=0 -> Q (B,H,S,D), z=1 -> K (B,H,S,D), z=2 -> V^T (B,H,D,S)
//    Grid (256,1,3): blockIdx.x = br*8 + bc (rows 4096/128, cols 1024/128).
// ---------------------------------------------------------------------------
__global__ __launch_bounds__(256) void proj_qkv_kernel(
        const u16* __restrict__ xb,
        const u16* __restrict__ wqT, const u16* __restrict__ wkT,
        const u16* __restrict__ wvT,
        u16* __restrict__ Qb, u16* __restrict__ Kb, u16* __restrict__ Vt) {
    const int tid = threadIdx.x;
    const int br = blockIdx.x >> 3;
    const int bc = blockIdx.x & 7;
    const int z  = blockIdx.z;
    const u16* wT = (z == 0) ? wqT : (z == 1) ? wkT : wvT;

    v8f acc[2][4];
#pragma unroll
    for (int t = 0; t < 2; ++t)
#pragma unroll
        for (int j = 0; j < 4; ++j)
            acc[t][j] = (v8f){0.f,0.f,0.f,0.f,0.f,0.f,0.f,0.f};

    gemm_tile_128(xb + (size_t)br * 128 * MODEL, wT + (size_t)bc * 128 * MODEL,
                  tid, acc);

    const int lane = tid & 31;
    const bool hi  = lane >= 16;
    const int l15  = lane & 15;
    const int wid  = tid >> 5;
    const int wr   = wid >> 1, wc = wid & 1;
#pragma unroll
    for (int t = 0; t < 2; ++t) {
#pragma unroll
        for (int j = 0; j < 4; ++j) {
#pragma unroll
            for (int r = 0; r < 8; ++r) {
                int row = br * 128 + wr * 32 + t * 16 + r + (hi ? 8 : 0);
                int col = bc * 128 + wc * 64 + j * 16 + l15;
                int b = row >> 11, s = row & (SEQ - 1);
                int h = col >> 6,  d = col & (HEAD_DIM - 1);
                u16 v = f32_to_bf16(acc[t][j][r]);
                size_t bh = (size_t)b * NUM_HEADS + h;
                if (z == 2)      Vt[(bh * HEAD_DIM + d) * SEQ + s] = v;
                else if (z == 0) Qb[(bh * SEQ + s) * HEAD_DIM + d] = v;
                else             Kb[(bh * SEQ + s) * HEAD_DIM + d] = v;
            }
        }
    }
}

// ---------------------------------------------------------------------------
// 4) Rep attention + per-tile scores + top-8 tile indices.
// ---------------------------------------------------------------------------
__global__ __launch_bounds__(256) void rep_topk_kernel(
        const u16* __restrict__ Qb, const u16* __restrict__ Kb,
        int* __restrict__ out_idx) {
    const int g = blockIdx.x & (NTILES - 1);
    const int h = (blockIdx.x >> 7) & (NUM_HEADS - 1);
    const int b = blockIdx.x >> 11;
    const int t = threadIdx.x;                       // 256 threads
    const size_t bh = (size_t)b * NUM_HEADS + h;
    const int rep = g * 16 + 15;

    __shared__ float qv[HEAD_DIM];
    __shared__ float pm[256];
    __shared__ float red[256];
    __shared__ float scores[NTILES];
    __shared__ int   topidx[TOPK];

    if (t < HEAD_DIM) qv[t] = bf16_to_f32(Qb[(bh * SEQ + rep) * HEAD_DIM + t]);
    __syncthreads();

    float lg[8];
    float pmax = NEGV;
    const float scale = 0.125f;                      // 1/sqrt(64)
    const int key0 = t * 8;
#pragma unroll
    for (int i = 0; i < 8; ++i) {
        int key = key0 + i;
        const uint4* kp4 = (const uint4*)(Kb + (bh * SEQ + key) * HEAD_DIM);
        float acc = 0.f;
#pragma unroll
        for (int c = 0; c < 8; ++c) {
            uint4 pk = kp4[c];
            const unsigned w[4] = {pk.x, pk.y, pk.z, pk.w};
#pragma unroll
            for (int e = 0; e < 4; ++e) {
                acc += qv[c * 8 + e * 2 + 0] * bf16_to_f32((u16)(w[e] & 0xffffu));
                acc += qv[c * 8 + e * 2 + 1] * bf16_to_f32((u16)(w[e] >> 16));
            }
        }
        float v = (key <= rep) ? acc * scale : NEGV;
        lg[i] = v;
        pmax = fmaxf(pmax, v);
    }
    pm[t] = pmax; red[t] = pmax;
    __syncthreads();
    for (int s = 128; s > 0; s >>= 1) {
        if (t < s) red[t] = fmaxf(red[t], red[t + s]);
        __syncthreads();
    }
    float m = red[0];
    __syncthreads();
    float ssum = 0.f;
#pragma unroll
    for (int i = 0; i < 8; ++i) ssum += __expf(lg[i] - m);
    red[t] = ssum;
    __syncthreads();
    for (int s = 128; s > 0; s >>= 1) {
        if (t < s) red[t] += red[t + s];
        __syncthreads();
    }
    float lsum = red[0];
    __syncthreads();
    if (t < NTILES) {
        float tmax = fmaxf(pm[2 * t], pm[2 * t + 1]);
        scores[t] = __expf(tmax - m) / lsum;         // monotone: score of tile max
    }
    __syncthreads();
    if (t == 0) {
#pragma unroll
        for (int j = 0; j < TOPK; ++j) {
            float best = -1.0e30f; int bi = 0;
            for (int i = 0; i < NTILES; ++i)
                if (scores[i] > best) { best = scores[i]; bi = i; }
            topidx[j] = bi;
            scores[bi] = -1.0e30f;
        }
    }
    __syncthreads();
    if (t < 16 * TOPK) {
        int s = g * 16 + (t >> 3);
        int j = t & (TOPK - 1);
        out_idx[(bh * SEQ + s) * TOPK + j] = topidx[j];
    }
}

// ---------------------------------------------------------------------------
// 5) Causal flash attention, transposed formulation.
//    One wave per (b,h, 16-query tile). S^T = K_tile x Q^T, O^T += V^T x P^T.
// ---------------------------------------------------------------------------
__global__ __launch_bounds__(256) void flash_attn_kernel(
        const u16* __restrict__ Qb, const u16* __restrict__ Kb,
        const u16* __restrict__ Vt, u16* __restrict__ attn) {
    const int lane = threadIdx.x & 31;
    const int wid  = threadIdx.x >> 5;
    const int task = blockIdx.x * 8 + wid;           // 4096 wave-tasks
    const int qt = task & (NTILES - 1);
    const int h  = (task >> 7) & (NUM_HEADS - 1);
    const int b  = task >> 11;
    const int q0 = qt * 16;
    const bool hi = lane >= 16;
    const int  l15 = lane & 15;
    const size_t bh = (size_t)b * NUM_HEADS + h;
    const float scale = 0.125f;

    // B-fragments of Q^T (d-chunks 0..31 and 32..63), loaded once
    ABFrag qB[2];
    const u16* qrow = Qb + (bh * SEQ + q0 + l15) * HEAD_DIM;
#pragma unroll
    for (int c = 0; c < 2; ++c) {
        const u16* p = qrow + c * 32 + (hi ? 8 : 0);
        qB[c].u[0] = *(const uint4*)(p);
        qB[c].u[1] = *(const uint4*)(p + 16);
    }

    v8f o[4];
#pragma unroll
    for (int j = 0; j < 4; ++j) o[j] = (v8f){0.f,0.f,0.f,0.f,0.f,0.f,0.f,0.f};
    float m = -INFINITY, l = 0.f;
    const int qcol = q0 + l15;
    const int kend = q0 + 16;                        // causal bound (exclusive)

    for (int k0 = 0; k0 < kend; k0 += 32) {
        // ---- logits S^T for 32 keys (two 16-key tiles) ----
        v8f st[2];
        st[0] = (v8f){0.f,0.f,0.f,0.f,0.f,0.f,0.f,0.f};
        st[1] = (v8f){0.f,0.f,0.f,0.f,0.f,0.f,0.f,0.f};
#pragma unroll
        for (int tkt = 0; tkt < 2; ++tkt) {
            const u16* krow = Kb + (bh * SEQ + k0 + tkt * 16 + l15) * HEAD_DIM;
#pragma unroll
            for (int c = 0; c < 2; ++c) {
                ABFrag ka;
                const u16* p = krow + c * 32 + (hi ? 8 : 0);
                ka.u[0] = *(const uint4*)(p);
                ka.u[1] = *(const uint4*)(p + 16);
                st[tkt] = __builtin_amdgcn_wmma_f32_16x16x32_bf16(
                    false, ka.v, false, qB[c].v, (short)0, st[tkt], false, false);
            }
        }
        // ---- mask + online softmax (per-lane = per-query-column state) ----
        float p0[8], p1[8];
        float mloc = NEGV;
#pragma unroll
        for (int r = 0; r < 8; ++r) {
            int key0 = k0 + r + (hi ? 8 : 0);
            int key1 = key0 + 16;
            float v0 = (key0 <= qcol) ? st[0][r] * scale : NEGV;
            float v1 = (key1 <= qcol) ? st[1][r] * scale : NEGV;
            p0[r] = v0; p1[r] = v1;
            mloc = fmaxf(mloc, fmaxf(v0, v1));
        }
        mloc = fmaxf(mloc, __shfl_xor(mloc, 16, 32));
        float m_new = fmaxf(m, mloc);
        float corr = __expf(m - m_new);
        float ls = 0.f;
#pragma unroll
        for (int r = 0; r < 8; ++r) {
            p0[r] = __expf(p0[r] - m_new);
            p1[r] = __expf(p1[r] - m_new);
            ls += p0[r] + p1[r];
        }
        ls += __shfl_xor(ls, 16, 32);
        l = l * corr + ls;
        m = m_new;
#pragma unroll
        for (int j = 0; j < 4; ++j) o[j] *= corr;
        // ---- P^T (C-layout) -> B-fragment (32 keys x 16 queries) ----
        ABFrag pB;
#pragma unroll
        for (int r = 0; r < 8; ++r) {
            float t0 = __shfl_xor(p0[r], 16, 32);
            float t1 = __shfl_xor(p1[r], 16, 32);
            float lo = hi ? t1 : p0[r];              // keys 0..7  / 16..23
            float hv = hi ? p1[r] : t0;              // keys 8..15 / 24..31
            pB.s[r]     = f32_to_bf16(lo);
            pB.s[r + 8] = f32_to_bf16(hv);
        }
        // ---- O^T += V^T x P^T ----
#pragma unroll
        for (int j = 0; j < 4; ++j) {
            ABFrag va;
            const u16* p = Vt + (bh * HEAD_DIM + j * 16 + l15) * SEQ + k0 + (hi ? 16 : 0);
            va.u[0] = *(const uint4*)(p);
            va.u[1] = *(const uint4*)(p + 8);
            o[j] = __builtin_amdgcn_wmma_f32_16x16x32_bf16(
                false, va.v, false, pB.v, (short)0, o[j], false, false);
        }
    }
    // ---- normalize + store (B,S,M) bf16 ----
    float inv_l = 1.f / l;
    int q = q0 + l15;
#pragma unroll
    for (int j = 0; j < 4; ++j) {
#pragma unroll
        for (int r = 0; r < 8; ++r) {
            int d = j * 16 + r + (hi ? 8 : 0);
            attn[((size_t)(b * SEQ + q)) * MODEL + h * HEAD_DIM + d] =
                f32_to_bf16(o[j][r] * inv_l);
        }
    }
}

// ---------------------------------------------------------------------------
// 6) out = attn(bf16) @ wo -> fp32 (LDS double-buffered WMMA GEMM).
// ---------------------------------------------------------------------------
__global__ __launch_bounds__(256) void out_gemm_kernel(
        const u16* __restrict__ attn, const u16* __restrict__ woT,
        float* __restrict__ out) {
    const int tid = threadIdx.x;
    const int br = blockIdx.x >> 3;
    const int bc = blockIdx.x & 7;

    v8f acc[2][4];
#pragma unroll
    for (int t = 0; t < 2; ++t)
#pragma unroll
        for (int j = 0; j < 4; ++j)
            acc[t][j] = (v8f){0.f,0.f,0.f,0.f,0.f,0.f,0.f,0.f};

    gemm_tile_128(attn + (size_t)br * 128 * MODEL, woT + (size_t)bc * 128 * MODEL,
                  tid, acc);

    const int lane = tid & 31;
    const bool hi  = lane >= 16;
    const int l15  = lane & 15;
    const int wid  = tid >> 5;
    const int wr   = wid >> 1, wc = wid & 1;
#pragma unroll
    for (int t = 0; t < 2; ++t)
#pragma unroll
        for (int j = 0; j < 4; ++j)
#pragma unroll
            for (int r = 0; r < 8; ++r) {
                int row = br * 128 + wr * 32 + t * 16 + r + (hi ? 8 : 0);
                int col = bc * 128 + wc * 64 + j * 16 + l15;
                out[(size_t)row * MODEL + col] = acc[t][j][r];
            }
}

// ---------------------------------------------------------------------------
// Host launch
// ---------------------------------------------------------------------------
extern "C" void kernel_launch(void* const* d_in, const int* in_sizes, int n_in,
                              void* d_out, int out_size, void* d_ws, size_t ws_size,
                              hipStream_t stream) {
    const float* x  = (const float*)d_in[0];
    const float* wq = (const float*)d_in[1];
    const float* wk = (const float*)d_in[2];
    const float* wv = (const float*)d_in[3];
    const float* wo = (const float*)d_in[4];

    float* outp = (float*)d_out;
    int*   idxp = (int*)(outp + (size_t)BATCH * SEQ * MODEL);

    // workspace layout (bytes)
    char* ws = (char*)d_ws;
    const size_t sz_xb  = (size_t)BATCH * SEQ * MODEL * 2;     // 8 MB
    const size_t sz_w   = (size_t)MODEL * MODEL * 2;           // 2 MB each
    const size_t sz_hsd = (size_t)BATCH * NUM_HEADS * SEQ * HEAD_DIM * 2; // 8 MB
    u16* xb   = (u16*)(ws);
    u16* wqT  = (u16*)(ws + sz_xb);
    u16* wkT  = (u16*)(ws + sz_xb + sz_w);
    u16* wvT  = (u16*)(ws + sz_xb + 2 * sz_w);
    u16* woT  = (u16*)(ws + sz_xb + 3 * sz_w);
    u16* Qb   = (u16*)(ws + sz_xb + 4 * sz_w);
    u16* Kb   = (u16*)(ws + sz_xb + 4 * sz_w + sz_hsd);
    u16* Vt   = (u16*)(ws + sz_xb + 4 * sz_w + 2 * sz_hsd);
    u16* attn = (u16*)(ws + sz_xb + 4 * sz_w + 3 * sz_hsd);

    const int nX = BATCH * SEQ * MODEL;
    cvt_x_kernel<<<(nX + 255) / 256, 256, 0, stream>>>(x, xb, nX);
    cvt_w_kernel<<<dim3(32, 32, 4), dim3(32, 8, 1), 0, stream>>>(
        wq, wk, wv, wo, wqT, wkT, wvT, woT);
    proj_qkv_kernel<<<dim3(256, 1, 3), 256, 0, stream>>>(
        xb, wqT, wkT, wvT, Qb, Kb, Vt);
    rep_topk_kernel<<<BATCH * NUM_HEADS * NTILES, 256, 0, stream>>>(Qb, Kb, idxp);
    flash_attn_kernel<<<512, 256, 0, stream>>>(Qb, Kb, Vt, attn);
    out_gemm_kernel<<<256, 256, 0, stream>>>(attn, woT, outp);
}
